// LabelWiseClassifier_34780645163188
// MI455X (gfx1250) — matrix operand order repaired and checked by reading, not compiled, not run on hardware
//
#include <hip/hip_runtime.h>

typedef __attribute__((ext_vector_type(16))) __bf16 v16bf;
typedef __attribute__((ext_vector_type(8)))  float  v8f;

#define HIDDEN 1024
#define HALFD  512
#define NCLS   128
#define BATCH  1024
#define NT     128   // N columns per workgroup (4 chunks cover 512)
#define KT     32    // K depth per LDS slab (= WMMA bf16 K)
#define LDSW   18    // dwords per LDS column row (36 halves: 32 K + pad; 8B aligned)
#define SLAB   (NT * LDSW)

#if defined(__has_builtin)
#if __has_builtin(__builtin_amdgcn_cvt_pk_bf16_f32)
#define HAVE_CVT_PK_BF16 1
#endif
#endif

__device__ __forceinline__ unsigned pack_bf16(float lo, float hi) {
#ifdef HAVE_CVT_PK_BF16
    auto t = __builtin_amdgcn_cvt_pk_bf16_f32(lo, hi);
    static_assert(sizeof(t) == 4, "cvt_pk_bf16 return size");
    unsigned r;
    __builtin_memcpy(&r, &t, 4);
    return r;
#else
    // round-half-up to bf16, then splice both high halves with one v_perm_b32:
    // 3 VALU per packed dword (vs 6 for bit-exact RNE)
    unsigned a = __float_as_uint(lo) + 0x8000u;
    unsigned b = __float_as_uint(hi) + 0x8000u;
    return __builtin_amdgcn_perm(b, a, 0x07060302u);  // {b[3],b[2],a[3],a[2]}
#endif
}

union AReg { v16bf v; unsigned u[8]; uint4 q[2]; };
union BReg { v16bf v; uint2 u2[4]; };

// ---- features f32 -> bf16 (one 2MB pass; features then served from L2 at half width)
__global__ void lwc_feat_to_bf16(const float* __restrict__ f,
                                 unsigned short* __restrict__ o) {
    int i = (blockIdx.x * 256 + threadIdx.x) * 8;
    float4 a = *(const float4*)(f + i);
    float4 b = *(const float4*)(f + i + 4);
    uint4 o4;
    o4.x = pack_bf16(a.x, a.y); o4.y = pack_bf16(a.z, a.w);
    o4.z = pack_bf16(b.x, b.y); o4.w = pack_bf16(b.z, b.w);
    *(uint4*)(o + i) = o4;
}

// ---- main fused kernel: per (class, N-chunk, M-half) GEMM + bias + ReLU + W2 dot
// grid = 1024 = class(128) x N-chunk(4) x M-half(2); block = 256 (8 wave32).
// Double-buffered LDS W1 slab + double-buffered A + preloaded B tiles:
// global and LDS latency both overlap the WMMA chain via partial counter waits.
template <bool PRECONV>
__global__ void __launch_bounds__(256)
lwc_gemm(const float* __restrict__ feat_f32,
         const unsigned short* __restrict__ feat_bf,
         const float* __restrict__ W1,
         const float* __restrict__ b1,
         const float* __restrict__ W2,
         float* __restrict__ part)
{
    __shared__ unsigned w1lds[2 * SLAB];   // 2 x 9216 B, bf16 column-major W1 slabs

    const int tid   = threadIdx.x;
    const int lane  = tid & 31;
    const int wv    = tid >> 5;
    const int c     = blockIdx.x >> 3;
    const int chunk = (blockIdx.x >> 1) & 3;
    const int mhalf = blockIdx.x & 1;
    const int n0    = chunk * NT;
    const int nl    = lane & 15;
    const int hi16  = lane >> 4;

    const float* W1c = W1 + (size_t)c * HIDDEN * HALFD + n0;

    // staging task: this thread owns K-rows 4*p2..4*p2+3, columns 4*g..4*g+3
    const int p2 = tid >> 5;
    const int g  = tid & 31;

    float b1v[8], w2v[8];
#pragma unroll
    for (int j = 0; j < 8; ++j) {
        b1v[j] = b1[(size_t)c * HALFD + n0 + j * 16 + nl];
        w2v[j] = W2[(size_t)c * HALFD + n0 + j * 16 + nl];
    }

    float4 s0, s1, s2, s3;          // staged W1 rows (global -> VGPR)
    auto stage_issue = [&](int k) {
        const float* r = W1c + (size_t)(k + 4 * p2) * HALFD + 4 * g;
        s0 = *(const float4*)r;
        s1 = *(const float4*)(r + HALFD);
        s2 = *(const float4*)(r + 2 * HALFD);
        s3 = *(const float4*)(r + 3 * HALFD);
    };

    AReg Aring[2];                  // A double buffer (PRECONV path)
    auto A_issue = [&](int itx) {
        const int rowA = mhalf * 512 + (itx >> 5) * NT + wv * 16 + nl;
        const unsigned short* ap =
            feat_bf + (size_t)rowA * HIDDEN + (itx & 31) * KT + (hi16 << 3);
        Aring[itx & 1].q[0] = *(const uint4*)ap;
        Aring[itx & 1].q[1] = *(const uint4*)(ap + 16);
    };

    v8f acc[8] = {};
    stage_issue(0);
    if (PRECONV) A_issue(0);

#pragma unroll 2                    // make it&1 parity static (regs stay in VGPRs)
    for (int it = 0; it < 128; ++it) {          // 4 M-tiles x 32 K-steps
        const int mt = it >> 5;
        const int kk = (it & 31) * KT;
        unsigned* buf = w1lds + (it & 1) * SLAB;

        // commit staged slab (VGPR -> LDS, transposed bf16, K-major per column)
        {
            const int ib = 2 * p2;
            *(uint2*)(buf + (4 * g + 0) * LDSW + ib) = uint2{pack_bf16(s0.x, s1.x), pack_bf16(s2.x, s3.x)};
            *(uint2*)(buf + (4 * g + 1) * LDSW + ib) = uint2{pack_bf16(s0.y, s1.y), pack_bf16(s2.y, s3.y)};
            *(uint2*)(buf + (4 * g + 2) * LDSW + ib) = uint2{pack_bf16(s0.z, s1.z), pack_bf16(s2.z, s3.z)};
            *(uint2*)(buf + (4 * g + 3) * LDSW + ib) = uint2{pack_bf16(s0.w, s1.w), pack_bf16(s2.w, s3.w)};
        }
        __syncthreads();

        // prefetch next W1 slab + next A tile (consumed next iteration)
        if (it + 1 < 128) {
            stage_issue(((it + 1) & 31) * KT);
            if (PRECONV) A_issue(it + 1);
        }

        AReg Acur;                  // fallback path: load+convert A in-iteration
        if (!PRECONV) {
            const int rowA = mhalf * 512 + mt * NT + wv * 16 + nl;
            const float* ap = feat_f32 + (size_t)rowA * HIDDEN + kk + (hi16 << 3);
            float4 f0 = *(const float4*)ap;
            float4 f1 = *(const float4*)(ap + 4);
            float4 f2 = *(const float4*)(ap + 16);
            float4 f3 = *(const float4*)(ap + 20);
            Acur.u[0] = pack_bf16(f0.x, f0.y); Acur.u[1] = pack_bf16(f0.z, f0.w);
            Acur.u[2] = pack_bf16(f1.x, f1.y); Acur.u[3] = pack_bf16(f1.z, f1.w);
            Acur.u[4] = pack_bf16(f2.x, f2.y); Acur.u[5] = pack_bf16(f2.z, f2.w);
            Acur.u[6] = pack_bf16(f3.x, f3.y); Acur.u[7] = pack_bf16(f3.z, f3.w);
        }

        // preload ALL 8 B tiles -> compiler ladders partial dscnt waits
        BReg B[8];
#pragma unroll
        for (int j = 0; j < 8; ++j) {
            const uint2* bp = (const uint2*)(buf + (j * 16 + nl) * LDSW + (hi16 << 2));
            B[j].u2[0] = bp[0]; B[j].u2[1] = bp[1];
            B[j].u2[2] = bp[4]; B[j].u2[3] = bp[5];
        }

        const v16bf Av = PRECONV ? Aring[it & 1].v : Acur.v;
#pragma unroll
        for (int j = 0; j < 8; ++j) {
            acc[j] = __builtin_amdgcn_wmma_f32_16x16x32_bf16(
                false, Av, false, B[j].v, (short)0, acc[j], false, false);
        }

        if ((it & 31) == 31) {
            // epilogue: relu(acc + b1) . w2 over this chunk's 128 N columns
            float ps[8];
#pragma unroll
            for (int r = 0; r < 8; ++r) {
                float p = 0.f;
#pragma unroll
                for (int j = 0; j < 8; ++j) {
                    float v = acc[j][r] + b1v[j];
                    v = v > 0.f ? v : 0.f;
                    p += v * w2v[j];
                }
#pragma unroll
                for (int m = 1; m < 16; m <<= 1)    // reduce 16-lane N groups
                    p += __shfl_xor(p, m, 16);
                ps[r] = p;
            }
            if (nl == 0) {
                const int rbase = mhalf * 512 + mt * NT + wv * 16 + (hi16 << 3);
#pragma unroll
                for (int r = 0; r < 8; ++r)
                    part[((size_t)chunk * BATCH + rbase + r) * NCLS + c] = ps[r];
            }
#pragma unroll
            for (int j = 0; j < 8; ++j) { v8f z = {}; acc[j] = z; }
        }
    }
}

// ---- deterministic cross-chunk reduction + b2
__global__ void lwc_reduce(const float* __restrict__ part,
                           const float* __restrict__ b2,
                           float* __restrict__ out)
{
    int idx = blockIdx.x * 256 + threadIdx.x;   // b*128 + c
    int c = idx & (NCLS - 1);
    float s = b2[c];
    s += part[idx];
    s += part[1 * BATCH * NCLS + idx];
    s += part[2 * BATCH * NCLS + idx];
    s += part[3 * BATCH * NCLS + idx];
    out[idx] = s;
}

extern "C" void kernel_launch(void* const* d_in, const int* in_sizes, int n_in,
                              void* d_out, int out_size, void* d_ws, size_t ws_size,
                              hipStream_t stream) {
    (void)in_sizes; (void)n_in; (void)out_size;
    const float* feat = (const float*)d_in[0];
    const float* W1   = (const float*)d_in[1];
    const float* b1   = (const float*)d_in[2];
    const float* W2   = (const float*)d_in[3];
    const float* b2   = (const float*)d_in[4];
    float* out = (float*)d_out;

    const size_t PART_BYTES   = (size_t)4 * BATCH * NCLS * sizeof(float); // 2 MB
    const size_t FEATBF_BYTES = (size_t)BATCH * HIDDEN * 2;               // 2 MB
    float* part = (float*)d_ws;
    unsigned short* fbf = (unsigned short*)((char*)d_ws + PART_BYTES);

    if (ws_size >= PART_BYTES + FEATBF_BYTES) {
        lwc_feat_to_bf16<<<BATCH * HIDDEN / (8 * 256), 256, 0, stream>>>(feat, fbf);
        lwc_gemm<true><<<1024, 256, 0, stream>>>(feat, fbf, W1, b1, W2, part);
    } else {
        lwc_gemm<false><<<1024, 256, 0, stream>>>(feat, nullptr, W1, b1, W2, part);
    }
    lwc_reduce<<<512, 256, 0, stream>>>(part, b2, out);
}
